// SwinTransformerBlock_16784732193424
// MI455X (gfx1250) — matrix operand order repaired and checked by reading, not compiled, not run on hardware
//
#include <hip/hip_runtime.h>
#include <hip/hip_bf16.h>
#include <math.h>

// ---------------------------------------------------------------------------
// SwinV2 block for MI455X (gfx1250): bf16 WMMA GEMMs + fp32 softmax/LN.
// B=32, H=W=56, DIM=384, NH=12, HD=32, WS=7, SHIFT=3, N=49, NW=64.
// M = B*NW*N = 100352 rows.
// ---------------------------------------------------------------------------

typedef __attribute__((ext_vector_type(16))) __bf16 v16bf;
typedef __attribute__((ext_vector_type(8)))  float  v8f;

#define WMMA_BF16(a, b, c) \
  __builtin_amdgcn_wmma_f32_16x16x32_bf16(false, (a), false, (b), (short)0, (c), false, false)

// Native bf16 converts (gfx1250 has hardware bf16 cvt ops; avoids the manual
// bfe/add3 RNE sequence that dominated VALU in the previous build).
static __device__ __forceinline__ unsigned short f2bf(float f) {
  __bf16 b = (__bf16)f;
  return __builtin_bit_cast(unsigned short, b);
}
static __device__ __forceinline__ float bf2f(unsigned short s) {
  return (float)__builtin_bit_cast(__bf16, s);
}
static __device__ __forceinline__ v8f zero8() {
  v8f z = {0.f, 0.f, 0.f, 0.f, 0.f, 0.f, 0.f, 0.f};
  return z;
}

// A fragment (16x32 bf16, 16 rows x K=32). Per ISA: lanes 0-15 -> M=lane,
// K runs [0..7] and [16..23]; lanes 16-31 -> M=lane-16, K runs [8..15],[24..31].
static __device__ __forceinline__ v16bf frag_load_a(const unsigned short* base, int ldk) {
  int lane = threadIdx.x & 31;
  int r = lane & 15;
  int lo = (lane >> 4) * 8;
  const unsigned short* p = base + r * ldk + lo;
  union { unsigned short us[16]; v16bf v; } u;
#pragma unroll
  for (int e = 0; e < 8; ++e) { u.us[e] = p[e]; u.us[8 + e] = p[16 + e]; }
  return u.v;
}

// B fragment (32x16 bf16, K=32 x 16 cols). Lanes 0-15: N=lane, K runs [0..7]
// and [16..23]; lanes 16-31: N=lane-16, K runs [8..15],[24..31]. LDS holds B as [n][k].
static __device__ __forceinline__ v16bf frag_load_b(const unsigned short* base, int ldk) {
  int lane = threadIdx.x & 31;
  const unsigned short* p = base + (lane & 15) * ldk + (lane >> 4) * 8;
  union { unsigned short us[16]; v16bf v; } u;
#pragma unroll
  for (int e = 0; e < 8; ++e) { u.us[e] = p[e]; u.us[8 + e] = p[16 + e]; }
  return u.v;
}

// ---------------------------------------------------------------------------
// Kernel 0a: CPB-MLP table  bt[i][h], i in 0..168 (13x13), h in 0..11
// ---------------------------------------------------------------------------
static __device__ __forceinline__ float rct_coord(int c) {
  float t = (float)c * (8.0f / 6.0f);
  float l = log2f(fabsf(t) + 1.0f) * (1.0f / 3.0f);
  return t < 0.f ? -l : l;
}

__global__ __launch_bounds__(128) void cpb_bt_kernel(
    const float* __restrict__ w1, const float* __restrict__ b1,
    const float* __restrict__ w2, float* __restrict__ bt) {
  int gid = blockIdx.x * 128 + threadIdx.x;
  if (gid >= 169 * 12) return;
  int i = gid / 12, h = gid - i * 12;
  int ih = i / 13, iw = i - ih * 13;
  float t0 = rct_coord(ih - 6);
  float t1 = rct_coord(iw - 6);
  float acc = 0.f;
  for (int j = 0; j < 512; ++j) {
    float hv = fmaxf(w1[2 * j] * t0 + w1[2 * j + 1] * t1 + b1[j], 0.f);
    acc += w2[h * 512 + j] * hv;
  }
  bt[gid] = acc;
}

// Kernel 0b: relb[h][n][m] = 16*sigmoid(bt[rpi(n,m)][h])
__global__ __launch_bounds__(128) void relbias_kernel(
    const float* __restrict__ bt, float* __restrict__ relb) {
  int gid = blockIdx.x * 128 + threadIdx.x;
  if (gid >= 12 * 49 * 49) return;
  int h = gid / 2401;
  int rem = gid - h * 2401;
  int n = rem / 49, m = rem - n * 49;
  int ny = n / 7, nx = n - ny * 7;
  int my = m / 7, mx = m - my * 7;
  int idx = (ny - my + 6) * 13 + (nx - mx + 6);
  float v = bt[idx * 12 + h];
  relb[gid] = 16.0f / (1.0f + expf(-v));
}

// ---------------------------------------------------------------------------
// Kernel 1: QKV GEMM (fused roll + window partition gather)
// out(100352 x 1152) = gather(x) @ qkv_w^T + qkv_bias, stored bf16 as
// q/k/v[(win*12+head)*64 + tok][hd]  (token padded to 64, pad never read)
// ---------------------------------------------------------------------------
__global__ __launch_bounds__(128) void qkv_gemm_kernel(
    const float* __restrict__ x, const float* __restrict__ qkv_w,
    const float* __restrict__ q_bias, const float* __restrict__ v_bias,
    unsigned short* __restrict__ q_ws, unsigned short* __restrict__ k_ws,
    unsigned short* __restrict__ v_ws) {
  __shared__ __align__(16) unsigned short As[64][40];
  __shared__ __align__(16) unsigned short Bs[64][40];
  int tid = threadIdx.x;
  int arow = tid >> 1, akoff = (tid & 1) * 16;
  // gather map for this A row (shift -3 roll + 7x7 window partition)
  int m = blockIdx.x * 64 + arow;
  int win = m / 49, tok = m - win * 49;
  int bimg = win >> 6, widx = win & 63;
  int wy = widx >> 3, wx = widx & 7;
  int ty = tok / 7, tx = tok - ty * 7;
  int hp = (wy * 7 + ty + 3) % 56;
  int wp = (wx * 7 + tx + 3) % 56;
  const float* asrc = x + ((size_t)bimg * 3136 + hp * 56 + wp) * 384 + akoff;
  int nglob = blockIdx.y * 64 + arow;
  const float* bsrc = qkv_w + (size_t)nglob * 384 + akoff;

  v8f acc[4];
#pragma unroll
  for (int i = 0; i < 4; ++i) acc[i] = zero8();

  for (int k0 = 0; k0 < 384; k0 += 32) {
    __syncthreads();
#pragma unroll
    for (int i = 0; i < 16; ++i) As[arow][akoff + i] = f2bf(asrc[k0 + i]);
#pragma unroll
    for (int i = 0; i < 16; ++i) Bs[arow][akoff + i] = f2bf(bsrc[k0 + i]);
    if (k0 + 32 < 384) __builtin_prefetch(asrc + k0 + 32, 0, 0);
    __syncthreads();
    int wv = tid >> 5;
    v16bf a = frag_load_a(&As[wv * 16][0], 40);
#pragma unroll
    for (int ct = 0; ct < 4; ++ct) {
      v16bf b = frag_load_b(&Bs[ct * 16][0], 40);
      acc[ct] = WMMA_BF16(a, b, acc[ct]);
    }
  }
  int wv = tid >> 5, lane = tid & 31;
  int hi = lane >> 4, lcol = lane & 15;
#pragma unroll
  for (int ct = 0; ct < 4; ++ct) {
    int n = blockIdx.y * 64 + ct * 16 + lcol;
    float bias = (n < 384) ? q_bias[n] : (n < 768 ? 0.f : v_bias[n - 768]);
    int t = n / 384, c = n - t * 384;
    int head = c >> 5, hd = c & 31;
    unsigned short* dst = (t == 0) ? q_ws : (t == 1 ? k_ws : v_ws);
#pragma unroll
    for (int r = 0; r < 8; ++r) {
      int mrow = blockIdx.x * 64 + wv * 16 + r + hi * 8;
      int w2 = mrow / 49, tk = mrow - w2 * 49;
      dst[(((size_t)w2 * 12 + head) * 64 + tk) * 32 + hd] = f2bf(acc[ct][r] + bias);
    }
  }
}

// ---------------------------------------------------------------------------
// Kernel 2: cosine attention per (window, head).
// ---------------------------------------------------------------------------
__global__ __launch_bounds__(128) void attn_kernel(
    const unsigned short* __restrict__ q_ws, const unsigned short* __restrict__ k_ws,
    const unsigned short* __restrict__ v_ws, const float* __restrict__ logit_scale,
    const float* __restrict__ relb, unsigned short* __restrict__ attn_out) {
  int win = blockIdx.x / 12;
  int h = blockIdx.x - win * 12;
  int tid = threadIdx.x;
  __shared__ __align__(16) unsigned short qn[64][40];
  __shared__ __align__(16) unsigned short kn[64][40];
  __shared__ __align__(16) unsigned short vT[32][72];
  __shared__ float S[64][65];
  __shared__ __align__(16) unsigned short P[64][72];
  __shared__ int cnts[64];

  size_t base = (((size_t)win * 12 + h) * 64) * 32;
  int widx = win & 63;
  int wy = widx >> 3, wx = widx & 7;
  if (tid < 64) {
    if (tid < 49) {
      int ty = tid / 7, tx = tid - ty * 7;
      int rh = wy * 7 + ty, cw = wx * 7 + tx;
      int gh = rh < 49 ? 0 : (rh < 53 ? 1 : 2);
      int gw = cw < 49 ? 0 : (cw < 53 ? 1 : 2);
      cnts[tid] = gh * 3 + gw;
    } else {
      cnts[tid] = -1;
    }
  }
  // normalize q (threads 0-63) and k (threads 64-127), zero-pad rows 49..63
  {
    int isK = tid >> 6;
    int row = tid & 63;
    unsigned short* dst = isK ? &kn[row][0] : &qn[row][0];
    if (row < 49) {
      const unsigned short* src = (isK ? k_ws : q_ws) + base + (size_t)row * 32;
      float vals[32];
      float ss = 0.f;
#pragma unroll
      for (int d = 0; d < 32; ++d) { float f = bf2f(src[d]); vals[d] = f; ss += f * f; }
      float sc = 1.0f / fmaxf(sqrtf(ss), 1e-12f);
#pragma unroll
      for (int d = 0; d < 32; ++d) dst[d] = f2bf(vals[d] * sc);
    } else {
#pragma unroll
      for (int d = 0; d < 32; ++d) dst[d] = 0;
    }
  }
  // v transposed [hd][token], zero-padded tokens 49..63 (pads WMMA K to 64)
  for (int s = tid; s < 32 * 64; s += 128) {
    int hd = s >> 6, tk = s & 63;
    vT[hd][tk] = (tk < 49) ? v_ws[base + (size_t)tk * 32 + hd] : (unsigned short)0;
  }
  __syncthreads();

  float scale = expf(fminf(logit_scale[h], 4.60517019f));  // ln(100)
  int wv = tid >> 5, lane = tid & 31;
  int hi = lane >> 4, lcol = lane & 15;

  // S = qn @ kn^T  (K = HD = 32, one WMMA per 16x16 tile)
  v16bf aq = frag_load_a(&qn[wv * 16][0], 40);
#pragma unroll
  for (int ct = 0; ct < 4; ++ct) {
    v16bf bk = frag_load_b(&kn[ct * 16][0], 40);
    v8f acc = zero8();
    acc = WMMA_BF16(aq, bk, acc);
    int mcol = ct * 16 + lcol;
#pragma unroll
    for (int r = 0; r < 8; ++r) {
      int nrow = wv * 16 + r + hi * 8;
      float sv;
      if (nrow < 49 && mcol < 49) {
        float mk = (cnts[nrow] != cnts[mcol]) ? -100.f : 0.f;
        sv = acc[r] * scale + relb[((size_t)h * 49 + nrow) * 49 + mcol] + mk;
      } else {
        sv = -1e9f;
      }
      S[nrow][mcol] = sv;
    }
  }
  __syncthreads();

  // fp32 softmax per row, write bf16 P (zero-padded cols/rows 49..63)
  if (tid < 64) {
    if (tid < 49) {
      float mx = -1e30f;
      for (int c = 0; c < 49; ++c) mx = fmaxf(mx, S[tid][c]);
      float sum = 0.f;
      for (int c = 0; c < 49; ++c) { float e = expf(S[tid][c] - mx); S[tid][c] = e; sum += e; }
      float inv = 1.0f / sum;
      for (int c = 0; c < 64; ++c)
        P[tid][c] = (c < 49) ? f2bf(S[tid][c] * inv) : (unsigned short)0;
    } else {
      for (int c = 0; c < 64; ++c) P[tid][c] = 0;
    }
  }
  __syncthreads();

  // O = P @ v   (K = tokens padded to 64 -> two K=32 WMMA steps, 2 col tiles)
  v8f oacc[2] = {zero8(), zero8()};
#pragma unroll
  for (int ks = 0; ks < 64; ks += 32) {
    v16bf ap = frag_load_a(&P[wv * 16][ks], 72);
#pragma unroll
    for (int ct = 0; ct < 2; ++ct) {
      v16bf bv = frag_load_b(&vT[ct * 16][ks], 72);
      oacc[ct] = WMMA_BF16(ap, bv, oacc[ct]);
    }
  }
#pragma unroll
  for (int ct = 0; ct < 2; ++ct) {
    int hd = ct * 16 + lcol;
#pragma unroll
    for (int r = 0; r < 8; ++r) {
      int tk = wv * 16 + r + hi * 8;
      if (tk < 49)
        attn_out[((size_t)win * 49 + tk) * 384 + h * 32 + hd] = f2bf(oacc[ct][r]);
    }
  }
}

// ---------------------------------------------------------------------------
// Kernel 3: proj GEMM + window-reverse + un-shift scatter (fp32 out)
// ---------------------------------------------------------------------------
__global__ __launch_bounds__(128) void proj_gemm_kernel(
    const unsigned short* __restrict__ A, const float* __restrict__ proj_w,
    const float* __restrict__ proj_b, float* __restrict__ y) {
  __shared__ __align__(16) unsigned short As[64][40];
  __shared__ __align__(16) unsigned short Bs[64][40];
  int tid = threadIdx.x;
  int arow = tid >> 1, akoff = (tid & 1) * 16;
  int m = blockIdx.x * 64 + arow;
  const unsigned short* asrc = A + (size_t)m * 384 + akoff;
  int nglob = blockIdx.y * 64 + arow;
  const float* bsrc = proj_w + (size_t)nglob * 384 + akoff;

  v8f acc[4];
#pragma unroll
  for (int i = 0; i < 4; ++i) acc[i] = zero8();

  for (int k0 = 0; k0 < 384; k0 += 32) {
    __syncthreads();
#pragma unroll
    for (int i = 0; i < 16; ++i) As[arow][akoff + i] = asrc[k0 + i];
#pragma unroll
    for (int i = 0; i < 16; ++i) Bs[arow][akoff + i] = f2bf(bsrc[k0 + i]);
    if (k0 + 32 < 384) __builtin_prefetch(asrc + k0 + 32, 0, 0);
    __syncthreads();
    int wv = tid >> 5;
    v16bf a = frag_load_a(&As[wv * 16][0], 40);
#pragma unroll
    for (int ct = 0; ct < 4; ++ct) {
      v16bf b = frag_load_b(&Bs[ct * 16][0], 40);
      acc[ct] = WMMA_BF16(a, b, acc[ct]);
    }
  }
  int wv = tid >> 5, lane = tid & 31;
  int hi = lane >> 4, lcol = lane & 15;
#pragma unroll
  for (int ct = 0; ct < 4; ++ct) {
    int n = blockIdx.y * 64 + ct * 16 + lcol;
    float bias = proj_b[n];
#pragma unroll
    for (int r = 0; r < 8; ++r) {
      int mrow = blockIdx.x * 64 + wv * 16 + r + hi * 8;
      int w2 = mrow / 49, tk = mrow - w2 * 49;
      int bimg = w2 >> 6, widx = w2 & 63;
      int wy = widx >> 3, wx = widx & 7;
      int ty = tk / 7, tx = tk - ty * 7;
      int hp = (wy * 7 + ty + 3) % 56;
      int wp = (wx * 7 + tx + 3) % 56;
      y[((size_t)bimg * 3136 + hp * 56 + wp) * 384 + n] = acc[ct][r] + bias;
    }
  }
}

// ---------------------------------------------------------------------------
// Kernel 4/7: out = resid + LayerNorm(y)*g + b   (one block per row of 384)
// ---------------------------------------------------------------------------
__global__ __launch_bounds__(128) void resid_ln_kernel(
    const float* __restrict__ resid, const float* __restrict__ y,
    const float* __restrict__ gamma, const float* __restrict__ beta,
    float* __restrict__ out) {
  int row = blockIdx.x;
  int tid = threadIdx.x;
  const float* yr = y + (size_t)row * 384;
  float v0 = yr[tid], v1 = yr[tid + 128], v2 = yr[tid + 256];
  __shared__ float red[128];
  red[tid] = v0 + v1 + v2;
  __syncthreads();
#pragma unroll
  for (int o = 64; o > 0; o >>= 1) {
    if (tid < o) red[tid] += red[tid + o];
    __syncthreads();
  }
  float mu = red[0] * (1.0f / 384.0f);
  __syncthreads();
  float d0 = v0 - mu, d1 = v1 - mu, d2 = v2 - mu;
  red[tid] = d0 * d0 + d1 * d1 + d2 * d2;
  __syncthreads();
#pragma unroll
  for (int o = 64; o > 0; o >>= 1) {
    if (tid < o) red[tid] += red[tid + o];
    __syncthreads();
  }
  float rstd = rsqrtf(red[0] * (1.0f / 384.0f) + 1e-5f);
  const float* rr = resid + (size_t)row * 384;
  float* orow = out + (size_t)row * 384;
  orow[tid]       = rr[tid]       + d0 * rstd * gamma[tid]       + beta[tid];
  orow[tid + 128] = rr[tid + 128] + d1 * rstd * gamma[tid + 128] + beta[tid + 128];
  orow[tid + 256] = rr[tid + 256] + d2 * rstd * gamma[tid + 256] + beta[tid + 256];
}

// ---------------------------------------------------------------------------
// Kernel 5: MLP GEMM1  h = gelu(x1 @ w1^T + b1), bf16 out (100352 x 1536)
// ---------------------------------------------------------------------------
__global__ __launch_bounds__(128) void mlp1_gemm_kernel(
    const float* __restrict__ x1, const float* __restrict__ w1,
    const float* __restrict__ b1, unsigned short* __restrict__ hmid) {
  __shared__ __align__(16) unsigned short As[64][40];
  __shared__ __align__(16) unsigned short Bs[64][40];
  int tid = threadIdx.x;
  int arow = tid >> 1, akoff = (tid & 1) * 16;
  int m = blockIdx.x * 64 + arow;
  const float* asrc = x1 + (size_t)m * 384 + akoff;
  int nglob = blockIdx.y * 64 + arow;
  const float* bsrc = w1 + (size_t)nglob * 384 + akoff;

  v8f acc[4];
#pragma unroll
  for (int i = 0; i < 4; ++i) acc[i] = zero8();

  for (int k0 = 0; k0 < 384; k0 += 32) {
    __syncthreads();
#pragma unroll
    for (int i = 0; i < 16; ++i) As[arow][akoff + i] = f2bf(asrc[k0 + i]);
#pragma unroll
    for (int i = 0; i < 16; ++i) Bs[arow][akoff + i] = f2bf(bsrc[k0 + i]);
    if (k0 + 32 < 384) __builtin_prefetch(asrc + k0 + 32, 0, 0);
    __syncthreads();
    int wv = tid >> 5;
    v16bf a = frag_load_a(&As[wv * 16][0], 40);
#pragma unroll
    for (int ct = 0; ct < 4; ++ct) {
      v16bf b = frag_load_b(&Bs[ct * 16][0], 40);
      acc[ct] = WMMA_BF16(a, b, acc[ct]);
    }
  }
  int wv = tid >> 5, lane = tid & 31;
  int hi = lane >> 4, lcol = lane & 15;
#pragma unroll
  for (int ct = 0; ct < 4; ++ct) {
    int n = blockIdx.y * 64 + ct * 16 + lcol;
    float bias = b1[n];
#pragma unroll
    for (int r = 0; r < 8; ++r) {
      int mrow = blockIdx.x * 64 + wv * 16 + r + hi * 8;
      float v = acc[ct][r] + bias;
      float g = 0.5f * v * (1.0f + erff(v * 0.70710678118f));  // exact GELU
      hmid[(size_t)mrow * 1536 + n] = f2bf(g);
    }
  }
}

// ---------------------------------------------------------------------------
// Kernel 6: MLP GEMM2  y2 = h @ w2^T + b2   (K = 1536, fp32 out)
// ---------------------------------------------------------------------------
__global__ __launch_bounds__(128) void mlp2_gemm_kernel(
    const unsigned short* __restrict__ hmid, const float* __restrict__ w2,
    const float* __restrict__ b2, float* __restrict__ y2) {
  __shared__ __align__(16) unsigned short As[64][40];
  __shared__ __align__(16) unsigned short Bs[64][40];
  int tid = threadIdx.x;
  int arow = tid >> 1, akoff = (tid & 1) * 16;
  int m = blockIdx.x * 64 + arow;
  const unsigned short* asrc = hmid + (size_t)m * 1536 + akoff;
  int nglob = blockIdx.y * 64 + arow;
  const float* bsrc = w2 + (size_t)nglob * 1536 + akoff;

  v8f acc[4];
#pragma unroll
  for (int i = 0; i < 4; ++i) acc[i] = zero8();

  for (int k0 = 0; k0 < 1536; k0 += 32) {
    __syncthreads();
#pragma unroll
    for (int i = 0; i < 16; ++i) As[arow][akoff + i] = asrc[k0 + i];
#pragma unroll
    for (int i = 0; i < 16; ++i) Bs[arow][akoff + i] = f2bf(bsrc[k0 + i]);
    if (k0 + 32 < 1536) __builtin_prefetch(asrc + k0 + 32, 0, 0);
    __syncthreads();
    int wv = tid >> 5;
    v16bf a = frag_load_a(&As[wv * 16][0], 40);
#pragma unroll
    for (int ct = 0; ct < 4; ++ct) {
      v16bf b = frag_load_b(&Bs[ct * 16][0], 40);
      acc[ct] = WMMA_BF16(a, b, acc[ct]);
    }
  }
  int wv = tid >> 5, lane = tid & 31;
  int hi = lane >> 4, lcol = lane & 15;
#pragma unroll
  for (int ct = 0; ct < 4; ++ct) {
    int n = blockIdx.y * 64 + ct * 16 + lcol;
    float bias = b2[n];
#pragma unroll
    for (int r = 0; r < 8; ++r) {
      int mrow = blockIdx.x * 64 + wv * 16 + r + hi * 8;
      y2[(size_t)mrow * 384 + n] = acc[ct][r] + bias;
    }
  }
}

// ---------------------------------------------------------------------------
// Launch: sequential on `stream`; workspace regions with lifetime aliasing.
//   regA (308.3 MB): qkv bf16 (K1->K2) | proj-out fp32 (K3->K4) | hmid bf16 (K5->K6)
//   regB (154.1 MB): attn_out bf16 (K2->K3) | y2 fp32 (K6->K7)
//   x1 fp32 (154.1 MB): K4 -> end.   Total ~617 MB.
// ---------------------------------------------------------------------------
extern "C" void kernel_launch(void* const* d_in, const int* in_sizes, int n_in,
                              void* d_out, int out_size, void* d_ws, size_t ws_size,
                              hipStream_t stream) {
  const float* x      = (const float*)d_in[0];
  const float* qkv_w  = (const float*)d_in[1];
  const float* q_bias = (const float*)d_in[2];
  const float* v_bias = (const float*)d_in[3];
  const float* lscale = (const float*)d_in[4];
  const float* cpb_w1 = (const float*)d_in[5];
  const float* cpb_b1 = (const float*)d_in[6];
  const float* cpb_w2 = (const float*)d_in[7];
  const float* proj_w = (const float*)d_in[8];
  const float* proj_b = (const float*)d_in[9];
  const float* n1g    = (const float*)d_in[10];
  const float* n1b    = (const float*)d_in[11];
  const float* n2g    = (const float*)d_in[12];
  const float* n2b    = (const float*)d_in[13];
  const float* mlp_w1 = (const float*)d_in[14];
  const float* mlp_b1 = (const float*)d_in[15];
  const float* mlp_w2 = (const float*)d_in[16];
  const float* mlp_b2 = (const float*)d_in[17];
  float* out = (float*)d_out;

  char* ws = (char*)d_ws;
  float* bt   = (float*)ws;                 // 169*12*4   = 8112 B   -> pad 8192
  float* relb = (float*)(ws + 8192);        // 12*49*49*4 = 115248 B -> pad 115456
  char* regA = ws + 8192 + 115456;
  unsigned short* q_ws = (unsigned short*)regA;           // 2048*12*64*32 bf16 each
  unsigned short* k_ws = q_ws + (size_t)50331648;
  unsigned short* v_ws = k_ws + (size_t)50331648;
  float* yb = (float*)regA;                                // proj out (aliases qkv)
  unsigned short* hmid = (unsigned short*)regA;            // MLP hidden (aliases)
  char* regB = regA + (size_t)308281344;
  unsigned short* attn_o = (unsigned short*)regB;          // 100352*384 bf16
  float* y2 = (float*)regB;                                // MLP out (aliases attn_o)
  float* x1 = (float*)(regB + (size_t)154140672);          // 100352*384 fp32

  cpb_bt_kernel<<<16, 128, 0, stream>>>(cpb_w1, cpb_b1, cpb_w2, bt);
  relbias_kernel<<<(12 * 49 * 49 + 127) / 128, 128, 0, stream>>>(bt, relb);
  qkv_gemm_kernel<<<dim3(1568, 18), 128, 0, stream>>>(x, qkv_w, q_bias, v_bias,
                                                      q_ws, k_ws, v_ws);
  attn_kernel<<<2048 * 12, 128, 0, stream>>>(q_ws, k_ws, v_ws, lscale, relb, attn_o);
  proj_gemm_kernel<<<dim3(1568, 6), 128, 0, stream>>>(attn_o, proj_w, proj_b, yb);
  resid_ln_kernel<<<100352, 128, 0, stream>>>(x, yb, n1g, n1b, x1);
  mlp1_gemm_kernel<<<dim3(1568, 24), 128, 0, stream>>>(x1, mlp_w1, mlp_b1, hmid);
  mlp2_gemm_kernel<<<dim3(1568, 6), 128, 0, stream>>>(hmid, mlp_w2, mlp_b2, y2);
  resid_ln_kernel<<<100352, 128, 0, stream>>>(x1, y2, n2g, n2b, out);
}
